// SHSABlock_CGLU_11235634446859
// MI455X (gfx1250) — matrix operand order repaired and checked by reading, not compiled, not run on hardware
//
#include <hip/hip_runtime.h>
#include <hip/hip_bf16.h>

#define BATCH   16
#define DIM     256
#define QK      16
#define PDIM    64
#define HID     341
#define HID2    682
#define NPIX    1024
#define QKV_OC  96
#define KPAD    32      // q/k feature dim padded 16 -> 32
#define FC2K    352     // fc2 K dim padded 341 -> 352 (multiple of 32)
#define FC1PAD  688     // fc1 output rows padded 682 -> 688

typedef __attribute__((ext_vector_type(16))) __bf16 v16bf;
typedef __attribute__((ext_vector_type(8)))  __bf16 v8bf;
typedef __attribute__((ext_vector_type(8)))  float  v8f;
typedef __attribute__((ext_vector_type(4)))  int    v4i;

// ---------- bf16 helpers (bit-exact RNE convert, no reliance on __bf16 arith) ----------
static __device__ __forceinline__ __bf16 f2bf(float f) {
  unsigned u = __builtin_bit_cast(unsigned, f);
  unsigned r = u + 0x7fffu + ((u >> 16) & 1u);
  unsigned short h = (unsigned short)(r >> 16);
  return __builtin_bit_cast(__bf16, h);
}
static __device__ __forceinline__ float bf2f(__bf16 h) {
  unsigned short s = __builtin_bit_cast(unsigned short, h);
  unsigned u = ((unsigned)s) << 16;
  return __builtin_bit_cast(float, u);
}

// ---------- CDNA5 async global->LDS copy (ASYNCcnt-tracked, no VGPR round trip) ----------
#if defined(__has_builtin)
#if __has_builtin(__builtin_amdgcn_global_load_async_to_lds_b128)
#define HAVE_ASYNC_LDS_BUILTIN 1
#endif
#if __has_builtin(__builtin_amdgcn_s_wait_asynccnt)
#define HAVE_WAIT_ASYNC_BUILTIN 1
#endif
#endif

// copy 16 bytes / lane from global to LDS, tracked by ASYNCcnt
static __device__ __forceinline__ void async_copy16(__bf16* lds_dst, const __bf16* gsrc) {
#if defined(HAVE_ASYNC_LDS_BUILTIN)
  __builtin_amdgcn_global_load_async_to_lds_b128(
      (__attribute__((address_space(1))) v4i*)(void*)gsrc,
      (__attribute__((address_space(3))) v4i*)(void*)lds_dst, 0, 0);
#else
  // VFLAT async form: VDST = LDS byte address, VADDR = 64-bit global address.
  // Low 32 bits of a generic LDS pointer are the LDS byte offset.
  unsigned loff = (unsigned)(unsigned long long)lds_dst;
  asm volatile("global_load_async_to_lds_b128 %0, %1, off"
               :: "v"(loff), "v"(gsrc) : "memory");
#endif
}
static __device__ __forceinline__ void wait_async0() {
#if defined(HAVE_WAIT_ASYNC_BUILTIN)
  __builtin_amdgcn_s_wait_asynccnt(0);
#else
  asm volatile("s_wait_asynccnt 0x0" ::: "memory");
#endif
}

// ---------- WMMA fragment loaders ----------
// A-matrix 16x32 bf16 layout: lane l holds row m=l&15; lane-half hi=l>>4 selects
// k in [kb+8*hi, +8) (VGPR0-3) and [kb+16+8*hi, +8) (VGPR4-7): two 16B loads.
static __device__ __forceinline__ v16bf frag_a_row(const __bf16* row, int kb, int lhi) {
  v8bf lo = *(const v8bf*)(row + kb + 8 * lhi);
  v8bf hi = *(const v8bf*)(row + kb + 16 + 8 * lhi);
  return __builtin_shufflevector(lo, hi, 0,1,2,3,4,5,6,7,8,9,10,11,12,13,14,15);
}
// B-matrix 32x16 bf16 layout: lane l holds column lr=l&15; k in [kb+8*hi, +8) and
// [kb+16+8*hi, +8): two 16B loads from the (column-major-per-output-channel) weight row.
static __device__ __forceinline__ v16bf frag_b_row(const __bf16* row, int kb, int lhi) {
  v8bf lo = *(const v8bf*)(row + kb + 8 * lhi);
  v8bf hi = *(const v8bf*)(row + kb + 16 + 8 * lhi);
  return __builtin_shufflevector(lo, hi, 0,1,2,3,4,5,6,7,8,9,10,11,12,13,14,15);
}
static __device__ __forceinline__ v8f wmma_bf16(v16bf a, v16bf b, v8f c) {
  return __builtin_amdgcn_wmma_f32_16x16x32_bf16(false, a, false, b, (short)0, c, false, false);
}
static __device__ __forceinline__ v8f zero8() {
  v8f z; for (int i = 0; i < 8; ++i) z[i] = 0.f; return z;
}

// ---------- K0: pack all 1x1-conv weights to bf16 [oc][K] (K padded, pad rows zero) ----------
__global__ void k_pack(const float* __restrict__ qkv_w, const float* __restrict__ proj_w,
                       const float* __restrict__ fc1_w, const float* __restrict__ fc2_w,
                       __bf16* __restrict__ qkvWb, __bf16* __restrict__ projWb,
                       __bf16* __restrict__ fc1Wb, __bf16* __restrict__ fc2Wb) {
  int tid = blockIdx.x * blockDim.x + threadIdx.x;
  int nth = gridDim.x * blockDim.x;
  for (int i = tid; i < QKV_OC * PDIM; i += nth) qkvWb[i] = f2bf(qkv_w[i]);
  for (int i = tid; i < DIM * DIM;    i += nth) projWb[i] = f2bf(proj_w[i]);
  for (int i = tid; i < FC1PAD * DIM; i += nth)
    fc1Wb[i] = (i < HID2 * DIM) ? f2bf(fc1_w[i]) : f2bf(0.f);
  for (int i = tid; i < DIM * FC2K;   i += nth) {
    int oc = i / FC2K, k = i % FC2K;
    fc2Wb[i] = (k < HID) ? f2bf(fc2_w[oc * HID + k]) : f2bf(0.f);
  }
}

// ---------- K1: depthwise 3x3 + bias + residual -> xA (NCHW fp32) ----------
__global__ void k_dwconv_res(const float* __restrict__ x, const float* __restrict__ w,
                             const float* __restrict__ bvec, float* __restrict__ xA) {
  int b = blockIdx.x >> 8;
  int c = blockIdx.x & 255;
  const float* xp = x  + ((size_t)(b * DIM + c)) * NPIX;
  float*       op = xA + ((size_t)(b * DIM + c)) * NPIX;
  __shared__ float t[34 * 34];
  for (int i = threadIdx.x; i < 34 * 34; i += blockDim.x) {
    int yy = i / 34 - 1, xx = i % 34 - 1;
    t[i] = (yy >= 0 && yy < 32 && xx >= 0 && xx < 32) ? xp[yy * 32 + xx] : 0.f;
  }
  __syncthreads();
  const float* wc = w + c * 9;
  float w0=wc[0],w1=wc[1],w2=wc[2],w3=wc[3],w4=wc[4],w5=wc[5],w6=wc[6],w7=wc[7],w8=wc[8];
  float bias = bvec[c];
  for (int p = threadIdx.x; p < NPIX; p += blockDim.x) {
    int py = p >> 5, px = p & 31;
    const float* tp = &t[py * 34 + px];
    float acc = bias + w0*tp[0] + w1*tp[1] + w2*tp[2]
                     + w3*tp[34] + w4*tp[35] + w5*tp[36]
                     + w6*tp[68] + w7*tp[69] + w8*tp[70];
    op[p] = xp[p] + acc;
  }
}

// ---------- K2: per-batch mean/rstd over first 64 channels ----------
__global__ void k_gnstats(const float* __restrict__ xA, float* __restrict__ stats) {
  int b = blockIdx.x;
  const float* p = xA + (size_t)b * DIM * NPIX;   // c<64 is the leading contiguous block
  float s = 0.f, q = 0.f;
  for (int i = threadIdx.x; i < PDIM * NPIX; i += blockDim.x) { float v = p[i]; s += v; q += v * v; }
  __shared__ float ss[256], qq[256];
  ss[threadIdx.x] = s; qq[threadIdx.x] = q; __syncthreads();
  for (int st = 128; st > 0; st >>= 1) {
    if (threadIdx.x < st) { ss[threadIdx.x] += ss[threadIdx.x + st]; qq[threadIdx.x] += qq[threadIdx.x + st]; }
    __syncthreads();
  }
  if (threadIdx.x == 0) {
    float inv = 1.f / (float)(PDIM * NPIX);
    float mu = ss[0] * inv;
    float var = qq[0] * inv - mu * mu;
    stats[2 * b] = mu; stats[2 * b + 1] = rsqrtf(var + 1e-5f);
  }
}

// ---------- K3: fused GroupNorm + qkv GEMM (K=64). q pre-scaled by QK^-0.5, K-dim zero-padded ----------
__global__ void k_qkv(const float* __restrict__ xA, const float* __restrict__ stats,
                      const float* __restrict__ gnw, const float* __restrict__ gnb,
                      const __bf16* __restrict__ qkvWb, const float* __restrict__ qkvb,
                      __bf16* __restrict__ Qp, __bf16* __restrict__ Kp, __bf16* __restrict__ Vt) {
  int b  = blockIdx.x >> 4;
  int n0 = (blockIdx.x & 15) * 64;
  float mu = stats[2 * b], rstd = stats[2 * b + 1];
  __shared__ __bf16 LA[64][64];                     // [local n][c] normalized bf16
  const float* xp = xA + (size_t)b * DIM * NPIX;
  for (int i = threadIdx.x; i < 64 * 64; i += blockDim.x) {
    int c = i >> 6, nn = i & 63;
    float v = xp[c * NPIX + n0 + nn];
    LA[nn][c] = f2bf((v - mu) * rstd * gnw[c] + gnb[c]);
  }
  __syncthreads();
  int lane = threadIdx.x & 31, w = threadIdx.x >> 5;
  int lr = lane & 15, lhi = lane >> 4;
  v8f acc[6]; for (int t = 0; t < 6; ++t) acc[t] = zero8();
  const __bf16* arow = &LA[16 * w + lr][0];
  for (int kb = 0; kb < PDIM; kb += 32) {
    v16bf af = frag_a_row(arow, kb, lhi);
    for (int t = 0; t < 6; ++t) {
      v16bf bfr = frag_b_row(qkvWb + (size_t)(16 * t + lr) * PDIM, kb, lhi);
      acc[t] = wmma_bf16(af, bfr, acc[t]);
    }
  }
  for (int t = 0; t < 6; ++t) {
    int oc = 16 * t + lr;
    float bias = qkvb[oc];
    for (int r = 0; r < 8; ++r) {
      int m = r + 8 * lhi;
      int n = n0 + 16 * w + m;
      float v = acc[t][r] + bias;
      if (t == 0) {                                  // q, fold softmax scale 0.25
        size_t bi = ((size_t)(b * NPIX + n)) * KPAD;
        Qp[bi + lr] = f2bf(v * 0.25f);
        Qp[bi + 16 + lr] = f2bf(0.f);
      } else if (t == 1) {                           // k
        size_t bi = ((size_t)(b * NPIX + n)) * KPAD;
        Kp[bi + lr] = f2bf(v);
        Kp[bi + 16 + lr] = f2bf(0.f);
      } else {                                       // v transposed: [b][c][n]
        int c = 16 * (t - 2) + lr;
        Vt[((size_t)(b * PDIM + c)) * NPIX + n] = f2bf(v);
      }
    }
  }
}

// ---------- K4: attention. block = 16 queries of one batch; scores strip in LDS ----------
__global__ void k_attn(const __bf16* __restrict__ Qp, const __bf16* __restrict__ Kp,
                       const __bf16* __restrict__ Vt, float* __restrict__ x1a) {
  int b  = blockIdx.x >> 6;
  int q0 = (blockIdx.x & 63) * 16;
  int tid = threadIdx.x;
  int lane = tid & 31, w = tid >> 5;
  int lr = lane & 15, lhi = lane >> 4;
  __shared__ __bf16 S[16][1024];                    // scores -> probabilities, in place
  __shared__ float red[16][8];
  __shared__ float rowstat[16];

  // Q fragment (row = query), reused across all 64 key tiles
  v16bf aq = frag_a_row(Qp + ((size_t)(b * NPIX + q0 + lr)) * KPAD, 0, lhi);
  for (int jt = 16 * w; jt < 16 * w + 16; ++jt) {
    int jb = jt * 16;
    v16bf bk = frag_b_row(Kp + ((size_t)(b * NPIX + jb + lr)) * KPAD, 0, lhi);
    v8f s = wmma_bf16(aq, bk, zero8());
    for (int r = 0; r < 8; ++r) S[r + 8 * lhi][jb + lr] = f2bf(s[r]);
  }
  __syncthreads();

  // softmax over 1024, 8 threads per row
  int row = tid >> 3, t0 = tid & 7;
  float mx = -1e30f;
  for (int c = t0; c < 1024; c += 8) mx = fmaxf(mx, bf2f(S[row][c]));
  red[row][t0] = mx; __syncthreads();
  if (t0 == 0) { float m = red[row][0]; for (int i = 1; i < 8; ++i) m = fmaxf(m, red[row][i]); rowstat[row] = m; }
  __syncthreads();
  float rm = rowstat[row], sum = 0.f;
  for (int c = t0; c < 1024; c += 8) { float e = __expf(bf2f(S[row][c]) - rm); S[row][c] = f2bf(e); sum += e; }
  red[row][t0] = sum; __syncthreads();
  if (t0 == 0) { float s2 = 0.f; for (int i = 0; i < 8; ++i) s2 += red[row][i]; rowstat[row] = 1.f / s2; }
  __syncthreads();
  float inv = rowstat[row];
  for (int c = t0; c < 1024; c += 8) S[row][c] = f2bf(bf2f(S[row][c]) * inv);
  __syncthreads();

  // PV: out(16 q x 16 c per wave) = P(16x1024) x V^T(1024x64), K-loop 32 steps
  int cb = 16 * w;
  v8f acc = zero8();
  const __bf16* arow = &S[lr][0];
  const __bf16* brow = Vt + ((size_t)(b * PDIM + cb + lr)) * NPIX;
  for (int kb = 0; kb < 1024; kb += 32) {
    v16bf ap = frag_a_row(arow, kb, lhi);            // ds_load_b128 x2
    v16bf bv = frag_b_row(brow, kb, lhi);
    acc = wmma_bf16(ap, bv, acc);
  }
  for (int r = 0; r < 8; ++r) {
    int m = r + 8 * lhi;
    x1a[((size_t)(b * PDIM + cb + lr)) * NPIX + q0 + m] = acc[r];
  }
}

// ---------- K5: concat(x1a, x2) -> SiLU -> Yb bf16 [b*n][256] ----------
__global__ void k_silu(const float* __restrict__ x1a, const float* __restrict__ xA,
                       __bf16* __restrict__ Yb) {
  int i = blockIdx.x * blockDim.x + threadIdx.x;
  int n = i & 1023;
  int c = (i >> 10) & 255;
  int b = i >> 18;
  float v = (c < PDIM) ? x1a[((size_t)(b * PDIM + c)) * NPIX + n]
                       : xA [((size_t)(b * DIM  + c)) * NPIX + n];
  float s = v / (1.f + __expf(-v));
  Yb[((size_t)(b * NPIX + n)) * DIM + c] = f2bf(s);
}

// ---------- K6: proj GEMM (K=256) + residual -> xB fp32 (NCHW) and Xb bf16 [row][256] ----------
// A tile (16 rows x 256 K = 8 KB, contiguous) staged once per block via async->LDS;
// all 4 waves then feed WMMA A-fragments from LDS instead of 4x-redundant global loads.
__global__ void k_proj(const __bf16* __restrict__ Yb, const __bf16* __restrict__ projWb,
                       const float* __restrict__ projb, const float* __restrict__ xA,
                       float* __restrict__ xB, __bf16* __restrict__ Xb) {
  int row0 = blockIdx.x * 16;
  int b = row0 >> 10, n0 = row0 & 1023;
  int lane = threadIdx.x & 31, w = threadIdx.x >> 5;
  int lr = lane & 15, lhi = lane >> 4;
  __shared__ __bf16 Atile[16 * DIM];                 // 8 KB
  {
    const __bf16* gsrc = Yb + (size_t)row0 * DIM;
    for (int it = 0; it < 4; ++it) {                 // 512 x 16B chunks / 128 threads
      int idx = threadIdx.x + it * 128;
      async_copy16(Atile + idx * 8, gsrc + idx * 8);
    }
    wait_async0();
  }
  __syncthreads();
  v8f acc[4]; for (int t = 0; t < 4; ++t) acc[t] = zero8();
  const __bf16* arow = Atile + lr * DIM;
  for (int kb = 0; kb < DIM; kb += 32) {
    v16bf af = frag_a_row(arow, kb, lhi);
    for (int t = 0; t < 4; ++t) {
      int oc0 = w * 64 + t * 16;
      const __bf16* brow = projWb + ((size_t)(oc0 + lr)) * DIM;
      __builtin_prefetch(brow + kb + 32, 0, 0);      // global_prefetch_b8 next k-step
      acc[t] = wmma_bf16(af, frag_b_row(brow, kb, lhi), acc[t]);
    }
  }
  for (int t = 0; t < 4; ++t) {
    int oc = w * 64 + t * 16 + lr;
    float bias = projb[oc];
    for (int r = 0; r < 8; ++r) {
      int m = r + 8 * lhi, n = n0 + m;
      size_t gi = ((size_t)(b * DIM + oc)) * NPIX + n;
      float v = acc[t][r] + bias + xA[gi];
      xB[gi] = v;
      Xb[((size_t)(row0 + m)) * DIM + oc] = f2bf(v);
    }
  }
}

// ---------- K7: fc1 GEMM (256 -> 682, padded 688) -> a,g fp32 (NCHW) ----------
__global__ void k_fc1(const __bf16* __restrict__ Xb, const __bf16* __restrict__ fc1Wb,
                      const float* __restrict__ fc1b,
                      float* __restrict__ abuf, float* __restrict__ gbuf) {
  int row0 = blockIdx.x * 16;
  int b = row0 >> 10, n0 = row0 & 1023;
  int lane = threadIdx.x & 31, w = threadIdx.x >> 5;
  int lr = lane & 15, lhi = lane >> 4;
  __shared__ __bf16 Atile[16 * DIM];                 // 8 KB
  {
    const __bf16* gsrc = Xb + (size_t)row0 * DIM;
    for (int it = 0; it < 4; ++it) {
      int idx = threadIdx.x + it * 128;
      async_copy16(Atile + idx * 8, gsrc + idx * 8);
    }
    wait_async0();
  }
  __syncthreads();
  int oc0 = blockIdx.y * 64 + w * 16;
  if (oc0 >= FC1PAD) return;                         // wave-uniform, after the barrier
  v8f acc = zero8();
  const __bf16* arow = Atile + lr * DIM;
  const __bf16* brow = fc1Wb + ((size_t)(oc0 + lr)) * DIM;
  for (int kb = 0; kb < DIM; kb += 32) {
    __builtin_prefetch(brow + kb + 32, 0, 0);
    acc = wmma_bf16(frag_a_row(arow, kb, lhi), frag_b_row(brow, kb, lhi), acc);
  }
  int oc = oc0 + lr;
  if (oc < HID2) {
    float bias = fc1b[oc];
    for (int r = 0; r < 8; ++r) {
      int n = n0 + r + 8 * lhi;
      float v = acc[r] + bias;
      if (oc < HID) abuf[((size_t)(b * HID + oc)) * NPIX + n] = v;
      else          gbuf[((size_t)(b * HID + (oc - HID))) * NPIX + n] = v;
    }
  }
}

// ---------- K8: depthwise 3x3 on a -> GELU(exact) -> * g -> Tb bf16 [b*n][352] (tail zeroed) ----------
__global__ void k_dwconv2(const float* __restrict__ abuf, const float* __restrict__ gbuf,
                          const float* __restrict__ w, const float* __restrict__ bvec,
                          __bf16* __restrict__ Tb) {
  int b = blockIdx.x / HID;
  int c = blockIdx.x % HID;
  const float* ap = abuf + ((size_t)(b * HID + c)) * NPIX;
  const float* gp = gbuf + ((size_t)(b * HID + c)) * NPIX;
  __shared__ float t[34 * 34];
  for (int i = threadIdx.x; i < 34 * 34; i += blockDim.x) {
    int yy = i / 34 - 1, xx = i % 34 - 1;
    t[i] = (yy >= 0 && yy < 32 && xx >= 0 && xx < 32) ? ap[yy * 32 + xx] : 0.f;
  }
  __syncthreads();
  const float* wc = w + c * 9;
  float w0=wc[0],w1=wc[1],w2=wc[2],w3=wc[3],w4=wc[4],w5=wc[5],w6=wc[6],w7=wc[7],w8=wc[8];
  float bias = bvec[c];
  for (int p = threadIdx.x; p < NPIX; p += blockDim.x) {
    int py = p >> 5, px = p & 31;
    const float* tp = &t[py * 34 + px];
    float acc = bias + w0*tp[0] + w1*tp[1] + w2*tp[2]
                     + w3*tp[34] + w4*tp[35] + w5*tp[36]
                     + w6*tp[68] + w7*tp[69] + w8*tp[70];
    float ge = 0.5f * acc * (1.f + erff(acc * 0.70710678118f));
    Tb[((size_t)(b * NPIX + p)) * FC2K + c] = f2bf(ge * gp[p]);
  }
  if (c < FC2K - HID) {                              // zero the K-pad columns
    for (int p = threadIdx.x; p < NPIX; p += blockDim.x)
      Tb[((size_t)(b * NPIX + p)) * FC2K + HID + c] = f2bf(0.f);
  }
}

// ---------- K9: fc2 GEMM (K=352) + bias + residual -> out (NCHW fp32) ----------
__global__ void k_fc2(const __bf16* __restrict__ Tb, const __bf16* __restrict__ fc2Wb,
                      const float* __restrict__ fc2b, const float* __restrict__ xB,
                      float* __restrict__ out) {
  int row0 = blockIdx.x * 16;
  int b = row0 >> 10, n0 = row0 & 1023;
  int lane = threadIdx.x & 31, w = threadIdx.x >> 5;
  int lr = lane & 15, lhi = lane >> 4;
  __shared__ __bf16 Atile[16 * FC2K];                // 11 KB, contiguous in global
  {
    const __bf16* gsrc = Tb + (size_t)row0 * FC2K;
    for (int idx = threadIdx.x; idx < (16 * FC2K) / 8; idx += 128)  // 704 x 16B chunks
      async_copy16(Atile + idx * 8, gsrc + idx * 8);
    wait_async0();
  }
  __syncthreads();
  v8f acc[4]; for (int t = 0; t < 4; ++t) acc[t] = zero8();
  const __bf16* arow = Atile + lr * FC2K;
  for (int kb = 0; kb < FC2K; kb += 32) {
    v16bf af = frag_a_row(arow, kb, lhi);
    for (int t = 0; t < 4; ++t) {
      int oc0 = w * 64 + t * 16;
      const __bf16* brow = fc2Wb + ((size_t)(oc0 + lr)) * FC2K;
      __builtin_prefetch(brow + kb + 32, 0, 0);
      acc[t] = wmma_bf16(af, frag_b_row(brow, kb, lhi), acc[t]);
    }
  }
  for (int t = 0; t < 4; ++t) {
    int oc = w * 64 + t * 16 + lr;
    float bias = fc2b[oc];
    for (int r = 0; r < 8; ++r) {
      int n = n0 + r + 8 * lhi;
      size_t gi = ((size_t)(b * DIM + oc)) * NPIX + n;
      out[gi] = acc[t][r] + bias + xB[gi];
    }
  }
}

extern "C" void kernel_launch(void* const* d_in, const int* in_sizes, int n_in,
                              void* d_out, int out_size, void* d_ws, size_t ws_size,
                              hipStream_t stream) {
  (void)in_sizes; (void)n_in; (void)out_size; (void)ws_size;
  const float* x      = (const float*)d_in[0];
  const float* conv_w = (const float*)d_in[1];
  const float* conv_b = (const float*)d_in[2];
  const float* gn_w   = (const float*)d_in[3];
  const float* gn_b   = (const float*)d_in[4];
  const float* qkv_w  = (const float*)d_in[5];
  const float* qkv_b  = (const float*)d_in[6];
  const float* proj_w = (const float*)d_in[7];
  const float* proj_b = (const float*)d_in[8];
  const float* fc1_w  = (const float*)d_in[9];
  const float* fc1_b  = (const float*)d_in[10];
  const float* dw_w   = (const float*)d_in[11];
  const float* dw_b   = (const float*)d_in[12];
  const float* fc2_w  = (const float*)d_in[13];
  const float* fc2_b  = (const float*)d_in[14];
  float* out = (float*)d_out;

  char* base = (char*)d_ws;
  size_t off = 0;
  auto alloc = [&](size_t bytes) -> char* {
    off = (off + 255) & ~(size_t)255;
    char* p = base + off;
    off += bytes;
    return p;
  };
  float*  xA     = (float*) alloc((size_t)BATCH * DIM  * NPIX * 4);   // ~16.8 MB
  float*  stats  = (float*) alloc((size_t)BATCH * 2 * 4);
  __bf16* Qp     = (__bf16*)alloc((size_t)BATCH * NPIX * KPAD * 2);   // 1 MB
  __bf16* Kp     = (__bf16*)alloc((size_t)BATCH * NPIX * KPAD * 2);   // 1 MB
  __bf16* Vt     = (__bf16*)alloc((size_t)BATCH * PDIM * NPIX * 2);   // 2 MB
  float*  x1a    = (float*) alloc((size_t)BATCH * PDIM * NPIX * 4);   // 4 MB
  __bf16* Yb     = (__bf16*)alloc((size_t)BATCH * NPIX * DIM  * 2);   // 8 MB
  float*  xB     = (float*) alloc((size_t)BATCH * DIM  * NPIX * 4);   // ~16.8 MB
  __bf16* Xb     = (__bf16*)alloc((size_t)BATCH * NPIX * DIM  * 2);   // 8 MB
  float*  abuf   = (float*) alloc((size_t)BATCH * HID  * NPIX * 4);   // ~22.3 MB
  float*  gbuf   = (float*) alloc((size_t)BATCH * HID  * NPIX * 4);   // ~22.3 MB
  __bf16* Tb     = (__bf16*)alloc((size_t)BATCH * NPIX * FC2K * 2);   // ~11.5 MB
  __bf16* qkvWb  = (__bf16*)alloc((size_t)QKV_OC * PDIM * 2);
  __bf16* projWb = (__bf16*)alloc((size_t)DIM * DIM * 2);
  __bf16* fc1Wb  = (__bf16*)alloc((size_t)FC1PAD * DIM * 2);
  __bf16* fc2Wb  = (__bf16*)alloc((size_t)DIM * FC2K * 2);

  k_pack      <<<256, 256, 0, stream>>>(qkv_w, proj_w, fc1_w, fc2_w, qkvWb, projWb, fc1Wb, fc2Wb);
  k_dwconv_res<<<BATCH * DIM, 256, 0, stream>>>(x, conv_w, conv_b, xA);
  k_gnstats   <<<BATCH, 256, 0, stream>>>(xA, stats);
  k_qkv       <<<BATCH * 16, 128, 0, stream>>>(xA, stats, gn_w, gn_b, qkvWb, qkv_b, Qp, Kp, Vt);
  k_attn      <<<BATCH * 64, 128, 0, stream>>>(Qp, Kp, Vt, x1a);
  k_silu      <<<(BATCH * DIM * NPIX) / 256, 256, 0, stream>>>(x1a, xA, Yb);
  k_proj      <<<BATCH * 64, 128, 0, stream>>>(Yb, projWb, proj_b, xA, xB, Xb);
  dim3 g7(BATCH * 64, (FC1PAD + 63) / 64);
  k_fc1       <<<g7, 128, 0, stream>>>(Xb, fc1Wb, fc1_b, abuf, gbuf);
  k_dwconv2   <<<BATCH * HID, 256, 0, stream>>>(abuf, gbuf, dw_w, dw_b, Tb);
  k_fc2       <<<BATCH * 64, 128, 0, stream>>>(Tb, fc2Wb, fc2_b, xB, out);
}
